// CNNMAD_65120294142148
// MI455X (gfx1250) — compile-verified, matrix operand order checked by reading, not run on hardware
//
#include <hip/hip_runtime.h>

// ---------------- problem constants ----------------
#define NX 1024
#define NY 1024
#define TT 512     // T
#define TPP 512    // TP
#define DD 128     // D
#define CC 8       // C
#define KC3 (TPP * DD)   // 65536, K of the big GEMM

typedef unsigned short u16;
typedef __attribute__((ext_vector_type(16))) __bf16 v16bf;
typedef __attribute__((ext_vector_type(8)))  float  v8f;

#define BM 128
#define BN 128
#define BK 32
#define PADK 40              // 32 bf16 + 8 pad = 80B = 20 dwords -> conflict-free b128
#define TILE_E (BM * PADK)   // elements per LDS tile (5120 u16)
#define TILE_B (TILE_E * 2)  // bytes per LDS tile (10240)

union FragU { uint4 q[2]; v16bf v; };

__device__ __forceinline__ u16 f2bf(float f) {
  unsigned u = __float_as_uint(f);
  unsigned r = u + 0x7FFFu + ((u >> 16) & 1u);   // round-to-nearest-even
  return (u16)(r >> 16);
}

// ---- fragment load from LDS matching CDNA5 16-bit A/B layouts ----
// lane L: row = base + (L&15); half = L>>4 selects K groups {0..7,16..23} / {8..15,24..31}
__device__ __forceinline__ v16bf load_frag(const u16* lds, int rowbase) {
  int lane = threadIdx.x & 31;
  int r    = rowbase + (lane & 15);
  int half = lane >> 4;
  const u16* p = lds + r * PADK + 8 * half;
  FragU u;
  u.q[0] = *(const uint4*)(p);        // K = 8*half + 0..7
  u.q[1] = *(const uint4*)(p + 16);   // K = 16 + 8*half + 0..7
  return u.v;
}

// ---- async-stage a 128 x 32 bf16 tile straight into LDS (no VGPR bounce) ----
// Each thread: 2 x global_load_async_to_lds_b128 (32B), 256 thr cover 8KB tile.
__device__ __forceinline__ void issue_async_tile(unsigned lds_base_b,
                                                 const u16* src, int ld, int ks) {
  int t = threadIdx.x;
  int r = t >> 1;                    // row 0..127
  int h = t & 1;                     // 32B half of the 64B row
  const u16* gp = src + (size_t)r * ld + ks + h * 16;
  unsigned lo = lds_base_b + (unsigned)(r * (PADK * 2) + h * 32);
  asm volatile(
      "global_load_async_to_lds_b128 %0, %1, off\n\t"
      "global_load_async_to_lds_b128 %0, %1, off offset:16"
      :: "v"(lo), "v"((unsigned long long)(uintptr_t)gp)
      : "memory");
}

// ---- shared WMMA core: double-buffered async pipeline ----
// 8 waves; each wave computes a 32(M) x 64(N) patch = 2x4 accumulator tiles.
__device__ __forceinline__ void gemm_core(const u16* Ag, int lda,
                                          const u16* Bg, int ldb,
                                          int K, u16* smem,
                                          v8f (&acc)[2][4]) {
  int w  = threadIdx.x >> 5;
  int wm = w & 3;          // M offset 32*wm
  int wn = w >> 2;         // N offset 64*wn
#pragma unroll
  for (int mi = 0; mi < 2; ++mi)
#pragma unroll
    for (int ni = 0; ni < 4; ++ni)
#pragma unroll
      for (int e = 0; e < 8; ++e) acc[mi][ni][e] = 0.0f;

  // LDS layout: [A0][A1][B0][B1], each TILE_B bytes
  unsigned base = (unsigned)(uintptr_t)(void*)smem;   // low 32 bits = LDS offset
  const int NS = K / BK;

  // prologue: stage 0 into buffer 0
  issue_async_tile(base + 0 * TILE_B, Ag, lda, 0);
  issue_async_tile(base + 2 * TILE_B, Bg, ldb, 0);

  for (int n = 0; n < NS; ++n) {
    int cur = n & 1;
    if (n + 1 < NS) {
      int nxt = (n + 1) & 1;
      issue_async_tile(base + (unsigned)(nxt)*TILE_B,       Ag, lda, (n + 1) * BK);
      issue_async_tile(base + (unsigned)(2 + nxt) * TILE_B, Bg, ldb, (n + 1) * BK);
      asm volatile("s_wait_asynccnt 0x4" ::: "memory");   // stage n's 4 loads done
    } else {
      asm volatile("s_wait_asynccnt 0x0" ::: "memory");
    }
    __syncthreads();                                      // stage n visible to all waves

    const u16* Acur = smem + (size_t)cur * TILE_E;
    const u16* Bcur = smem + (size_t)(2 + cur) * TILE_E;
    v16bf af[2], bfr[4];
#pragma unroll
    for (int mi = 0; mi < 2; ++mi) af[mi]  = load_frag(Acur, wm * 32 + mi * 16);
#pragma unroll
    for (int ni = 0; ni < 4; ++ni) bfr[ni] = load_frag(Bcur, wn * 64 + ni * 16);
#pragma unroll
    for (int mi = 0; mi < 2; ++mi)
#pragma unroll
      for (int ni = 0; ni < 4; ++ni)
        acc[mi][ni] = __builtin_amdgcn_wmma_f32_16x16x32_bf16(
            false, af[mi], false, bfr[ni], (short)0, acc[mi][ni], false, false);
    __syncthreads();   // done reading buf[cur] before stage n+2 overwrites it
  }
}

// ================= small prep kernels =================

__global__ void k_rowsum(const float* __restrict__ pi, float* __restrict__ rowC) {
  int tid = blockIdx.x * 256 + threadIdx.x;        // C*T = 4096
  int c = tid >> 9, t = tid & 511;
  const float* p = pi + ((size_t)c * TT + t) * TPP;
  float s = 0.f;
  for (int q = 0; q < TPP; ++q) s += p[q];
  rowC[tid] = s;
}

__global__ void k_colsum(const float* __restrict__ pi, float* __restrict__ colC) {
  int tid = blockIdx.x * 256 + threadIdx.x;        // C*TP = 4096
  int c = tid >> 9, q = tid & 511;
  float s = 0.f;
  for (int t = 0; t < TT; ++t) s += pi[((size_t)c * TT + t) * TPP + q];
  colC[tid] = s;
}

// piT[c][p][t] = bf16(pi[c][t][p])
__global__ void k_pit(const float* __restrict__ pi, u16* __restrict__ piT) {
  int tid = blockIdx.x * 256 + threadIdx.x;        // C*TP*T = 2M
  int t = tid & 511;
  int rest = tid >> 9;
  int p = rest & 511;
  int c = rest >> 9;
  piT[tid] = f2bf(pi[((size_t)c * TT + t) * TPP + p]);
}

// XTb[i][d][t] = bf16(X[i][t][d])  (LDS-tiled transpose, 128x128 tiles)
__global__ __launch_bounds__(256) void k_xtb(const float* __restrict__ X,
                                             u16* __restrict__ XTb) {
  __shared__ u16 tile[128 * 132];                  // pad 132 -> conflict-free col reads
  int i  = blockIdx.y;
  int t0 = blockIdx.x * 128;
#pragma unroll 4
  for (int it = 0; it < 64; ++it) {
    int idx = threadIdx.x + it * 256;
    int r = idx >> 7, d = idx & 127;
    tile[r * 132 + d] = f2bf(X[((size_t)i * TT + t0 + r) * DD + d]);
  }
  __syncthreads();
#pragma unroll 4
  for (int it = 0; it < 64; ++it) {
    int idx = threadIdx.x + it * 256;
    int d = idx >> 7, tl = idx & 127;
    XTb[((size_t)i * DD + d) * TT + t0 + tl] = tile[tl * 132 + d];
  }
}

__global__ __launch_bounds__(256) void k_c1(const float* __restrict__ X,
                                            const float* __restrict__ rowC,
                                            const int* __restrict__ classe,
                                            float* __restrict__ C1v) {
  int i = blockIdx.x;
  int cl = classe[i];
  const float* Xi = X + (size_t)i * TT * DD;
  const float* rc = rowC + cl * TT;
  float s = 0.f;
  for (int idx = threadIdx.x; idx < TT * DD; idx += 256) {
    float x = Xi[idx];
    s += x * x * rc[idx >> 7];
  }
  __shared__ float red[256];
  red[threadIdx.x] = s;
  __syncthreads();
  for (int o = 128; o > 0; o >>= 1) {
    if ((int)threadIdx.x < o) red[threadIdx.x] += red[threadIdx.x + o];
    __syncthreads();
  }
  if (threadIdx.x == 0) C1v[i] = red[0];
}

// YS[j][p] = sum_d Y[j][p][d]^2  (one wave per row, coalesced)
__global__ __launch_bounds__(256) void k_ys(const float* __restrict__ Y,
                                            float* __restrict__ YS) {
  int j = blockIdx.x;
  int w = threadIdx.x >> 5, lane = threadIdx.x & 31;
  for (int p = w; p < TPP; p += 8) {
    const float* r = Y + ((size_t)j * TPP + p) * DD;
    float s = 0.f;
    for (int d = lane; d < DD; d += 32) { float y = r[d]; s += y * y; }
    for (int o = 16; o > 0; o >>= 1) s += __shfl_xor(s, o, 32);
    if (lane == 0) YS[(size_t)j * TPP + p] = s;
  }
}

__global__ void k_c2(const float* __restrict__ colC, const float* __restrict__ YS,
                     float* __restrict__ C2v) {
  int tid = blockIdx.x * 256 + threadIdx.x;        // C*NY = 8192
  int c = tid >> 10, j = tid & 1023;
  const float* cc = colC + c * TPP;
  const float* ys = YS + (size_t)j * TPP;
  float s = 0.f;
  for (int p = 0; p < TPP; ++p) s += cc[p] * ys[p];
  C2v[tid] = s;
}

__global__ void k_yb(const float* __restrict__ Y, u16* __restrict__ Yb) {
  size_t idx = (size_t)blockIdx.x * 256 + threadIdx.x;   // NY*TP*D/4 threads
  float4 v = ((const float4*)Y)[idx];
  ushort4 o;
  o.x = f2bf(v.x); o.y = f2bf(v.y); o.z = f2bf(v.z); o.w = f2bf(v.w);
  ((ushort4*)Yb)[idx] = o;
}

// ================= GEMM kernels =================

// XP[i] = piT[cl(i)] @ X_i   -> XPb bf16 [NX][TP][D]
__global__ __launch_bounds__(256) void k_xp(const u16* __restrict__ piT,
                                            const u16* __restrict__ XTb,
                                            const int* __restrict__ classe,
                                            u16* __restrict__ XPb) {
  __shared__ __align__(16) u16 smem[4 * TILE_E];   // A0 A1 B0 B1
  int i  = blockIdx.y;
  int p0 = blockIdx.x * BM;
  int cl = classe[i];
  v8f acc[2][4];
  gemm_core(piT + ((size_t)cl * TPP + p0) * TT, TT,
            XTb + (size_t)i * DD * TT, TT, TT, smem, acc);
  int lane = threadIdx.x & 31;
  int w = threadIdx.x >> 5, wm = w & 3, wn = w >> 2;
  int half = lane >> 4, nl = lane & 15;
  u16* out = XPb + (size_t)i * TPP * DD;
#pragma unroll
  for (int mi = 0; mi < 2; ++mi)
#pragma unroll
    for (int v = 0; v < 8; ++v) {
      int p = p0 + wm * 32 + mi * 16 + half * 8 + v;
#pragma unroll
      for (int ni = 0; ni < 4; ++ni) {
        int d = wn * 64 + ni * 16 + nl;
        out[(size_t)p * DD + d] = f2bf(acc[mi][ni][v]);
      }
    }
}

// out[i][j] = C1[i] + C2[cl(i)][j] - 2 * (XPb[i] . Yb[j])
__global__ __launch_bounds__(256) void k_c3(const u16* __restrict__ XPb,
                                            const u16* __restrict__ Yb,
                                            const float* __restrict__ C1v,
                                            const float* __restrict__ C2v,
                                            const int* __restrict__ classe,
                                            float* __restrict__ out) {
  __shared__ __align__(16) u16 smem[4 * TILE_E];   // A0 A1 B0 B1
  int j0 = blockIdx.x * BN;
  int i0 = blockIdx.y * BM;
  v8f acc[2][4];
  gemm_core(XPb + (size_t)i0 * KC3, KC3,
            Yb  + (size_t)j0 * KC3, KC3, KC3, smem, acc);
  int lane = threadIdx.x & 31;
  int w = threadIdx.x >> 5, wm = w & 3, wn = w >> 2;
  int half = lane >> 4, nl = lane & 15;
#pragma unroll
  for (int mi = 0; mi < 2; ++mi)
#pragma unroll
    for (int v = 0; v < 8; ++v) {
      int i = i0 + wm * 32 + mi * 16 + half * 8 + v;
      float c1 = C1v[i];
      const float* c2 = C2v + (size_t)classe[i] * NY;
#pragma unroll
      for (int ni = 0; ni < 4; ++ni) {
        int j = j0 + wn * 64 + ni * 16 + nl;
        out[(size_t)i * NY + j] = c1 + c2[j] - 2.0f * acc[mi][ni][v];
      }
    }
}

// ================= launcher =================
extern "C" void kernel_launch(void* const* d_in, const int* in_sizes, int n_in,
                              void* d_out, int out_size, void* d_ws, size_t ws_size,
                              hipStream_t stream) {
  const float* X      = (const float*)d_in[0];   // [NX,T,D]
  const float* Y      = (const float*)d_in[1];   // [NY,TP,D]
  const float* pi     = (const float*)d_in[2];   // [C,T,TP]
  const int*   classe = (const int*)d_in[3];     // [NX]
  float* out = (float*)d_out;                    // [NX,NY]

  char* w = (char*)d_ws;
  const size_t SZBIG = (size_t)NX * TPP * DD * sizeof(u16);   // 128 MB
  u16* XPb  = (u16*)w;                       // [NX][TP][D] bf16
  u16* BUF2 = (u16*)(w + SZBIG);             // first XTb [NX][D][T], later Yb [NY][TP*D]
  u16* piT  = (u16*)(w + 2 * SZBIG);         // [C][TP][T] bf16 (4 MB)
  float* rowC = (float*)(w + 2 * SZBIG + ((size_t)CC * TPP * TT * sizeof(u16)));
  float* colC = rowC + CC * TT;              // [C][TP]
  float* C1v  = colC + CC * TPP;             // [NX]
  float* C2v  = C1v + NX;                    // [C][NY]
  float* YS   = C2v + (size_t)CC * NY;       // [NY][TP]

  // --- prep ---
  k_rowsum<<<CC * TT / 256, 256, 0, stream>>>(pi, rowC);
  k_colsum<<<CC * TPP / 256, 256, 0, stream>>>(pi, colC);
  k_pit<<<(CC * TPP * TT) / 256, 256, 0, stream>>>(pi, piT);
  k_xtb<<<dim3(TT / 128, NX), 256, 0, stream>>>(X, BUF2);
  k_c1<<<NX, 256, 0, stream>>>(X, rowC, classe, C1v);
  k_ys<<<NY, 256, 0, stream>>>(Y, YS);
  k_c2<<<(CC * NY) / 256, 256, 0, stream>>>(colC, YS, C2v);

  // --- GEMM 1: XP (uses XTb in BUF2) ---
  k_xp<<<dim3(TPP / BM, NX), 256, 0, stream>>>(piT, BUF2, classe, XPb);

  // --- convert Y -> bf16, reusing BUF2 (XTb no longer needed) ---
  k_yb<<<((size_t)NY * TPP * DD / 4) / 256, 256, 0, stream>>>(Y, BUF2);

  // --- GEMM 2: C3 + fused epilogue ---
  k_c3<<<dim3(NY / BN, NX / BM), 256, 0, stream>>>(XPb, BUF2, C1v, C2v, classe, out);
}
